// AxialAttention_55070070669939
// MI455X (gfx1250) — matrix operand order repaired and checked by reading, not compiled, not run on hardware
//
#include <hip/hip_runtime.h>

// ---------------------------------------------------------------------------
// AxialAttention for MI455X (gfx1250, wave32, WMMA).
//   * QKV projection: BN folded into bf16 weights/bias, run as per-image GEMM
//     M=1024 x N=3136 (h*W+w, contiguous!) x K=512 with
//     V_WMMA_F32_16X16X32_BF16, B tile staged through LDS (coalesced loads,
//     no masking anywhere -> no exec-mask branching in the hot loop).
//   * Attention: one block per (batch-column, head); qk and attn@v on WMMA,
//     i-dependent relative-embedding contractions on VALU, softmax via wave32
//     shuffles, all through LDS.
// Memory roofline: ~450 MB HBM traffic -> ~19 us at 23.3 TB/s; bf16 matrix
// pipes keep the 63 GFLOP of compute well under that.
// ---------------------------------------------------------------------------

typedef __attribute__((ext_vector_type(16))) __bf16 v16bf;
typedef __attribute__((ext_vector_type(8)))  float  v8f;

#define EPSF 0.001f
constexpr int Nn = 16, Cc = 512, Hh = 56, Ww = 56;
constexpr int Gg = 8;          // heads
constexpr int CHq = 32;        // q / k channels per head
constexpr int CVv = 64;        // v channels per head
constexpr int Dd = 1024;       // 2*C qkv channels
constexpr int HP = 64;         // H padded to 64 for WMMA tiling
constexpr int Bb = Nn * Ww;    // 896 independent attention columns
constexpr int HW = Hh * Ww;    // 3136 = GEMM N per image (196 exact 16-tiles)

// A-fragment K index for 16-bit 16x32 A matrix (ISA 7.12.2):
// lanes 0-15: VGPR0-3 -> K{0..7}, VGPR4-7 -> K{16..23}; lanes 16-31: +8.
__device__ __forceinline__ int a_k_of(int e, int kSel) {
  int v = e >> 1;
  return (v < 4) ? (kSel + 2 * v + (e & 1))
                 : (16 + kSel + 2 * (v - 4) + (e & 1));
}

// ---------------------------------------------------------------------------
// Kernel 1: fold qkv-BN into weights/bias (bf16), expand rel_emb -> all_emb.
// ---------------------------------------------------------------------------
__global__ __launch_bounds__(256) void prep_kernel(
    const float* __restrict__ wq, const float* __restrict__ qg,
    const float* __restrict__ qb, const float* __restrict__ qm,
    const float* __restrict__ qv, const float* __restrict__ rel,
    __bf16* __restrict__ w_bf, float* __restrict__ bias,
    float* __restrict__ all_emb) {
  int i = blockIdx.x * blockDim.x + threadIdx.x;
  if (i < Dd * Cc) {
    int d = i >> 9;  // / 512
    float s = qg[d] * rsqrtf(qv[d] + EPSF);
    w_bf[i] = (__bf16)(wq[i] * s);
  }
  if (i < Dd) {
    float s = qg[i] * rsqrtf(qv[i] + EPSF);
    bias[i] = qb[i] - qm[i] * s;
  }
  if (i < 2 * CVv * HW) {  // 128 * 3136
    int cidx = i / HW;
    int rem = i - cidx * HW;
    int ii = rem / Hh, jj = rem - ii * Hh;
    all_emb[i] = rel[cidx * (2 * Hh - 1) + (ii - jj + Hh - 1)];
  }
}

// ---------------------------------------------------------------------------
// Kernel 2: qkv projection GEMM.
//   Per image n: D(1024) x HW(3136) x C(512), columns hw = h*56+w are
//   CONTIGUOUS in x -> coalesced B loads, and 3136 % 16 == 0 -> no masks.
//   Output layout [n][d][w][h(64)] so the attention kernel reads h-contiguous.
//   Grid (16, 196, 8), block 256 = 8 waves, each owning a 16-row M stripe.
// ---------------------------------------------------------------------------
__global__ __launch_bounds__(256) void qkv_gemm_kernel(
    const float* __restrict__ x,      // [N][C][3136]
    const __bf16* __restrict__ wq,    // [1024][512] BN-folded
    const float* __restrict__ bias,   // [1024]
    __bf16* __restrict__ qkv) {       // [N][1024][56][64]
  __shared__ __bf16 xs[16][32];       // B tile, transposed: [hw-col][k]

  const int n = blockIdx.x;
  const int hw0 = blockIdx.y * 16;
  const int dt = blockIdx.z;
  const int tid = threadIdx.x;
  const int wave = tid >> 5, lane = tid & 31;
  const int ml = lane & 15;

  const int m0 = dt * 128 + wave * 16;     // this wave's d-row base
  const int kSelA = (lane < 16) ? 0 : 8;   // A K-group select
  const int kSelB = (lane < 16) ? 0 : 16;  // B K-group select

  const __bf16* arow0 = wq + (size_t)(m0 + ml) * Cc;
  const float* xn = x + (size_t)n * Cc * HW + hw0;

  // cooperative-staging coords: 512 elements, 2 consecutive columns/thread
  const int cS = tid >> 3;           // k row 0..31
  const int uS = (tid * 2) & 15;     // hw col 0,2,..,14

  v8f acc = {};
  for (int k0 = 0; k0 < Cc; k0 += 32) {
    // ---- stage x[k0..k0+31][hw0..hw0+15] -> LDS (bf16, transposed) ----
    {
      const float* p = xn + (size_t)(k0 + cS) * HW + uS;
      float x0 = p[0];
      float x1 = p[1];
      xs[uS][cS] = (__bf16)x0;
      xs[uS + 1][cS] = (__bf16)x1;
    }
    __syncthreads();

    if (k0 + 32 < Cc) __builtin_prefetch((const void*)(arow0 + k0 + 32), 0, 3);

    v16bf af, bfv;
#pragma unroll
    for (int e = 0; e < 16; ++e) af[e] = arow0[k0 + a_k_of(e, kSelA)];
#pragma unroll
    for (int e = 0; e < 16; ++e) bfv[e] = xs[ml][a_k_of(e, kSelA)];  // 2x16B LDS runs
    acc = __builtin_amdgcn_wmma_f32_16x16x32_bf16(
        false, af, false, bfv, (short)0, acc, false, false);
    __syncthreads();
  }

  // D layout: lanes 0-15 VGPR r -> M=r, N=lane; lanes 16-31 -> M=8+r.
  const int hw = hw0 + ml;
  const int h = hw / Ww, w = hw - h * Ww;
#pragma unroll
  for (int r = 0; r < 8; ++r) {
    int m = m0 + ((lane < 16) ? r : 8 + r);
    float val = acc[r] + bias[m];
    qkv[((size_t)(n * Dd + m) * Ww + w) * HP + h] = (__bf16)val;
  }
}

// ---------------------------------------------------------------------------
// Kernel 3: per-(b, head) attention block. Grid (896, 8), block 256 (8 waves).
// ---------------------------------------------------------------------------
__global__ __launch_bounds__(256) void attn_kernel(
    const __bf16* __restrict__ qkv,   // [N][1024][56][64]
    const float* __restrict__ all_emb,
    const float* __restrict__ sg, const float* __restrict__ sb,
    const float* __restrict__ sm, const float* __restrict__ svr,
    const float* __restrict__ og, const float* __restrict__ ob,
    const float* __restrict__ om, const float* __restrict__ ov,
    float* __restrict__ out) {
  __shared__ __bf16 q_s[CHq][HP];
  __shared__ __bf16 k_s[CHq][HP];
  __shared__ __bf16 v_s[CVv][HP];
  __shared__ float att[HP][HP];    // logits -> attention weights
  __shared__ float sv_s[CVv][HP];  // attn @ v^T result

  const int b = blockIdx.x, g = blockIdx.y;
  const int n = b / Ww, w = b - n * Ww;
  const int wave = threadIdx.x >> 5, lane = threadIdx.x & 31;

  // Folded stacked-BN scalars (channels: qk = g, qr = 8+g, kr = 16+g)
  const float a_qk = sg[g] * rsqrtf(svr[g] + EPSF);
  const float a_qr = sg[8 + g] * rsqrtf(svr[8 + g] + EPSF);
  const float a_kr = sg[16 + g] * rsqrtf(svr[16 + g] + EPSF);
  const float bsum = (sb[g] - sm[g] * a_qk) + (sb[8 + g] - sm[8 + g] * a_qr) +
                     (sb[16 + g] - sm[16 + g] * a_kr);

  // ---- stage q,k,v into LDS; source is h-contiguous, pad h>=56 with 0 ----
  const __bf16* src = qkv + ((size_t)(n * Dd + g * 128) * Ww + w) * HP;
  for (int idx = threadIdx.x; idx < 128 * HP; idx += 256) {
    int cc = idx >> 6, hh = idx & 63;
    __bf16 val = (hh < Hh) ? src[(size_t)cc * (Ww * HP) + hh] : (__bf16)0.0f;
    if (cc < CHq)          q_s[cc][hh] = val;
    else if (cc < 2 * CHq) k_s[cc - CHq][hh] = val;
    else                   v_s[cc - 2 * CHq][hh] = val;
  }
  __syncthreads();

  // ---- logits: qk via WMMA (K=32 in one step) + VALU qr/kr, fused BN ----
  for (int t = wave; t < 16; t += 8) {
    const int ti = t >> 2, tj = t & 3;
    const int i0 = ti * 16, j0 = tj * 16;
    const int ml = lane & 15;
    const int kSelA = (lane < 16) ? 0 : 8;
    const int kSelB = (lane < 16) ? 0 : 16;
    v16bf af, bfv;
#pragma unroll
    for (int e = 0; e < 16; ++e) af[e] = q_s[a_k_of(e, kSelA)][i0 + ml];
#pragma unroll
    for (int e = 0; e < 16; ++e) bfv[e] = k_s[a_k_of(e, kSelA)][j0 + ml];
    v8f accq = {};
    accq = __builtin_amdgcn_wmma_f32_16x16x32_bf16(
        false, af, false, bfv, (short)0, accq, false, false);
#pragma unroll
    for (int r = 0; r < 8; ++r) {
      const int ii = i0 + ((lane < 16) ? r : 8 + r);
      const int jj = j0 + ml;
      float val = -3.0e38f;  // mask padded columns out of the softmax
      if (ii < Hh && jj < Hh) {
        float qr = 0.f, kr = 0.f;
        const float* qe = all_emb + (size_t)ii * Hh + jj;                    // q_emb[c][ii][jj]
        const float* ke = all_emb + (size_t)CHq * HW + (size_t)jj * Hh + ii; // k_emb[c][jj][ii]
        for (int cc = 0; cc < CHq; ++cc) {
          qr += (float)q_s[cc][ii] * qe[(size_t)cc * HW];
          kr += (float)k_s[cc][jj] * ke[(size_t)cc * HW];
        }
        val = a_qk * accq[r] + a_qr * qr + a_kr * kr + bsum;
      }
      att[ii][jj] = val;
    }
  }
  __syncthreads();

  // ---- row softmax (wave32: two elements per lane, shfl_xor reductions) ----
  for (int i = wave; i < Hh; i += 8) {
    float v0 = att[i][lane];
    float v1 = att[i][lane + 32];
    float mx = fmaxf(v0, v1);
#pragma unroll
    for (int off = 16; off > 0; off >>= 1) mx = fmaxf(mx, __shfl_xor(mx, off));
    float e0 = __expf(v0 - mx);
    float e1 = __expf(v1 - mx);
    float s = e0 + e1;
#pragma unroll
    for (int off = 16; off > 0; off >>= 1) s += __shfl_xor(s, off);
    float inv = 1.0f / s;
    att[i][lane] = e0 * inv;
    att[i][lane + 32] = e1 * inv;
  }
  // zero padded rows so sv WMMA padding stays clean
  for (int i = Hh + wave; i < HP; i += 8) {
    att[i][lane] = 0.f;
    att[i][lane + 32] = 0.f;
  }
  __syncthreads();

  // ---- sv[c,i] = sum_j v[c,j] * attn[i,j] via WMMA, K=64 (2 steps) ----
  for (int t = wave; t < 16; t += 8) {
    const int tc = t >> 2, ti = t & 3;
    const int ml = lane & 15;
    const int m = tc * 16 + ml;   // c row
    const int ni = ti * 16 + ml;  // i column
    const int kSelA = (lane < 16) ? 0 : 8;
    const int kSelB = (lane < 16) ? 0 : 16;
    v8f acc = {};
#pragma unroll
    for (int k0 = 0; k0 < HP; k0 += 32) {
      v16bf af, bfv;
#pragma unroll
      for (int e = 0; e < 16; ++e) af[e] = v_s[m][k0 + a_k_of(e, kSelA)];
#pragma unroll
      for (int e = 0; e < 16; ++e) bfv[e] = (__bf16)att[ni][k0 + a_k_of(e, kSelA)];
      acc = __builtin_amdgcn_wmma_f32_16x16x32_bf16(
          false, af, false, bfv, (short)0, acc, false, false);
    }
#pragma unroll
    for (int r = 0; r < 8; ++r)
      sv_s[tc * 16 + ((lane < 16) ? r : 8 + r)][ti * 16 + ml] = acc[r];
  }
  __syncthreads();

  // ---- sve (i-dependent, VALU) + output BN + NCHW transpose store ----
  const float* ve_base = all_emb + (size_t)(2 * CHq) * HW;  // v_emb[64][56][56]
  for (int idx = threadIdx.x; idx < CVv * Hh; idx += 256) {
    const int c = idx / Hh, i = idx - c * Hh;
    const float* ve = ve_base + ((size_t)c * Hh + i) * Hh;
    float s = 0.f;
    for (int j = 0; j < Hh; ++j) s += att[i][j] * ve[j];
    float y = sv_s[c][i] + s;
    const int Cg = g * CVv + c;
    const float sc = og[Cg] * rsqrtf(ov[Cg] + EPSF);
    y = (y - om[Cg]) * sc + ob[Cg];
    out[(((size_t)n * Cc + Cg) * Hh + i) * Ww + w] = y;
  }
}

// ---------------------------------------------------------------------------
extern "C" void kernel_launch(void* const* d_in, const int* in_sizes, int n_in,
                              void* d_out, int out_size, void* d_ws,
                              size_t ws_size, hipStream_t stream) {
  (void)in_sizes; (void)n_in; (void)out_size; (void)ws_size;
  const float* x   = (const float*)d_in[0];
  const float* wq  = (const float*)d_in[1];
  const float* qg  = (const float*)d_in[2];
  const float* qb  = (const float*)d_in[3];
  const float* qm  = (const float*)d_in[4];
  const float* qv  = (const float*)d_in[5];
  const float* rel = (const float*)d_in[6];
  const float* sg  = (const float*)d_in[7];
  const float* sb  = (const float*)d_in[8];
  const float* sm  = (const float*)d_in[9];
  const float* sv  = (const float*)d_in[10];
  const float* og  = (const float*)d_in[11];
  const float* ob  = (const float*)d_in[12];
  const float* om  = (const float*)d_in[13];
  const float* ov  = (const float*)d_in[14];
  float* out = (float*)d_out;

  // Workspace layout
  const size_t qkv_bytes = (size_t)Nn * Dd * Ww * HP * sizeof(__bf16);  // 117,440,512
  const size_t w_bytes   = (size_t)Dd * Cc * sizeof(__bf16);            //   1,048,576
  const size_t b_bytes   = (size_t)Dd * sizeof(float);                  //       4,096
  char* ws = (char*)d_ws;
  __bf16* qkv_bf  = (__bf16*)ws;
  __bf16* w_bf    = (__bf16*)(ws + qkv_bytes);
  float*  bias    = (float*)(ws + qkv_bytes + w_bytes);
  float*  all_emb = (float*)(ws + qkv_bytes + w_bytes + b_bytes);

  prep_kernel<<<(Dd * Cc + 255) / 256, 256, 0, stream>>>(
      wq, qg, qb, qm, qv, rel, w_bf, bias, all_emb);

  dim3 g2(Nn, HW / 16, Dd / 128);  // (16, 196, 8)
  qkv_gemm_kernel<<<g2, 256, 0, stream>>>(x, w_bf, bias, qkv_bf);

  dim3 g3(Bb, Gg);  // (896, 8)
  attn_kernel<<<g3, 256, 0, stream>>>(qkv_bf, all_emb, sg, sb, sm, sv, og, ob,
                                      om, ov, out);
}
